// Eyettention_41369124995125
// MI455X (gfx1250) — matrix operand order repaired and checked by reading, not compiled, not run on hardware
//
#include <hip/hip_runtime.h>
#include <hip/hip_bf16.h>

// ---------------------------------------------------------------------------
// Eyettention decode: 8-layer LSTM stack + windowed attention + MLP head.
// bf16 WMMA (v_wmma_f32_16x16x32_bf16), 32x32 per-wave register tile (2x2
// fragments -> 4 WMMAs per 8 b128 loads), fp32 accumulate/state.
// ---------------------------------------------------------------------------

typedef __attribute__((ext_vector_type(16))) __bf16 v16bf;
typedef __attribute__((ext_vector_type(8)))  float  v8f;
typedef __attribute__((ext_vector_type(4)))  float  f32x4;

#define B_ 1024
#define S_ 27
#define T_ 51
#define H_ 128

__device__ __forceinline__ unsigned short f2bf(float f) {
  unsigned int u = __float_as_uint(f);
  unsigned int r = u + 0x7FFFu + ((u >> 16) & 1u);   // round-to-nearest-even
  return (unsigned short)(r >> 16);
}
__device__ __forceinline__ float bf2f(unsigned short h) {
  return __uint_as_float(((unsigned int)h) << 16);
}
__device__ __forceinline__ float sigm(float x) { return 1.0f / (1.0f + __expf(-x)); }

// ------------------------------ prep kernels -------------------------------

__global__ void zero_u16_kernel(unsigned short* p, int n) {
  int i = blockIdx.x * blockDim.x + threadIdx.x;
  if (i < n) p[i] = 0;
}
__global__ void zero_f32_kernel(float* p, int n) {
  int i = blockIdx.x * blockDim.x + threadIdx.x;
  if (i < n) p[i] = 0.0f;
}
// Copy a [rows, copyK] fp32 block into bf16 dst at column offset dstOff, ld dstLd.
__global__ void pack_sub_kernel(const float* __restrict__ src, unsigned short* __restrict__ dst,
                                int rows, int srcK, int dstLd, int dstOff, int copyK) {
  int i = blockIdx.x * blockDim.x + threadIdx.x;
  int total = rows * copyK;
  if (i >= total) return;
  int n = i / copyK, k = i - n * copyK;
  dst[(size_t)n * dstLd + dstOff + k] = f2bf(src[(size_t)n * srcK + k]);
}
__global__ void bias_comb_kernel(const float* __restrict__ bi, const float* __restrict__ bh,
                                 float* __restrict__ out, int n) {
  int i = blockIdx.x * blockDim.x + threadIdx.x;
  if (i < n) out[i] = bi[i] + bh[i];
}
// dec_emb_in[t] fp32 [B,770] -> bf16 xin0[:, 0:770] (ld 928)
__global__ void convert_x_kernel(const float* __restrict__ x, unsigned short* __restrict__ xin0, int t) {
  int i = blockIdx.x * blockDim.x + threadIdx.x;
  if (i >= B_ * 770) return;
  int b = i / 770, k = i - b * 770;
  xin0[(size_t)b * 928 + k] = f2bf(x[(size_t)t * B_ * 770 + i]);
}

// ------------------------------ WMMA GEMM ----------------------------------
// Y[M, N] = X[M, Kpad] @ W[Npad, Kpad]^T + bias ; optional relu;
// writes fp32 (ldYf) and/or bf16 (ldYb).
// 32x32 output tile per wave (2x2 WMMA fragments), 8 waves per block.
// Mt2 = M/32, Nt2 = Npad/32; grid = Mt2*Nt2/8 blocks of 256 threads.

union AFrag { v16bf v; f32x4 f[2]; };

__global__ void wmma_gemm_kernel(const unsigned short* __restrict__ X, int ldX,
                                 const unsigned short* __restrict__ W, int ldW,
                                 const float* __restrict__ bias,
                                 float* __restrict__ Yf, int ldYf,
                                 unsigned short* __restrict__ Yb, int ldYb,
                                 int Mt2, int Nt2, int Nvalid, int Kpad, int relu) {
  const int wave = threadIdx.x >> 5;
  const int lane = threadIdx.x & 31;
  const int tile = blockIdx.x * 8 + wave;
  const int mT = tile % Mt2;
  const int nT = tile / Mt2;
  const int g = lane >> 4;      // half-lane-group
  const int r = lane & 15;      // A: row m ; B: col n
  const unsigned short* Xr0 = X + (size_t)(mT * 32 + r) * ldX;
  const unsigned short* Xr1 = Xr0 + (size_t)16 * ldX;
  const unsigned short* Wr0 = W + (size_t)(nT * 32 + r) * ldW;
  const unsigned short* Wr1 = Wr0 + (size_t)16 * ldW;
  v8f acc00 = {}, acc01 = {}, acc10 = {}, acc11 = {};
  for (int k = 0; k < Kpad; k += 32) {
    AFrag a0, a1, b0, b1;
    // A 16x32 bf16 layout: VGPR0-3 hold K = g*8 + 0..7 ; VGPR4-7 hold K = 16 + g*8 + 0..7
    a0.f[0] = *(const f32x4*)(Xr0 + k + g * 8);
    a0.f[1] = *(const f32x4*)(Xr0 + k + 16 + g * 8);
    a1.f[0] = *(const f32x4*)(Xr1 + k + g * 8);
    a1.f[1] = *(const f32x4*)(Xr1 + k + 16 + g * 8);
    // B 32x16 bf16 layout: same K order as A (g*8 + 0..7, then 16 + g*8 + 0..7)
    b0.f[0] = *(const f32x4*)(Wr0 + k + g * 8);
    b0.f[1] = *(const f32x4*)(Wr0 + k + 16 + g * 8);
    b1.f[0] = *(const f32x4*)(Wr1 + k + g * 8);
    b1.f[1] = *(const f32x4*)(Wr1 + k + 16 + g * 8);
    if (k + 32 < Kpad) {
      __builtin_prefetch(Xr0 + k + 32, 0, 1);
      __builtin_prefetch(Wr0 + k + 32, 0, 1);
    }
    acc00 = __builtin_amdgcn_wmma_f32_16x16x32_bf16(false, a0.v, false, b0.v,
                                                    (short)0, acc00, false, false);
    acc01 = __builtin_amdgcn_wmma_f32_16x16x32_bf16(false, a0.v, false, b1.v,
                                                    (short)0, acc01, false, false);
    acc10 = __builtin_amdgcn_wmma_f32_16x16x32_bf16(false, a1.v, false, b0.v,
                                                    (short)0, acc10, false, false);
    acc11 = __builtin_amdgcn_wmma_f32_16x16x32_bf16(false, a1.v, false, b1.v,
                                                    (short)0, acc11, false, false);
  }
  // Store: divergence only after all WMMAs are done (EXEC all-ones rule).
  const int col0 = nT * 32 + r;
  const int col1 = col0 + 16;
  const float bc0 = (col0 < Nvalid && bias) ? bias[col0] : 0.0f;
  const float bc1 = (col1 < Nvalid && bias) ? bias[col1] : 0.0f;
#pragma unroll
  for (int i = 0; i < 2; ++i) {
    const v8f* ai = i ? &acc10 : &acc00;   // col0 accumulators for row-half i
    const v8f* bi2 = i ? &acc11 : &acc01;  // col1 accumulators for row-half i
#pragma unroll
    for (int v = 0; v < 8; ++v) {          // C layout: row = v + g*8 within tile
      int row = mT * 32 + i * 16 + g * 8 + v;
      float v0 = (*ai)[v] + bc0;
      float v1 = (*bi2)[v] + bc1;
      if (relu) { v0 = fmaxf(v0, 0.0f); v1 = fmaxf(v1, 0.0f); }
      if (col0 < Nvalid) {
        if (Yf) Yf[(size_t)row * ldYf + col0] = v0;
        if (Yb) Yb[(size_t)row * ldYb + col0] = f2bf(v0);
      }
      if (col1 < Nvalid) {
        if (Yf) Yf[(size_t)row * ldYf + col1] = v1;
        if (Yb) Yb[(size_t)row * ldYb + col1] = f2bf(v1);
      }
    }
  }
}

// --------------------------- LSTM gate pointwise ---------------------------
// gates [B,512] (i,f,g,o) -> update c (fp32), write h (bf16) to recurrent slot
// and to next consumer's input slot.
__global__ void lstm_gate_kernel(const float* __restrict__ gates, float* __restrict__ c,
                                 unsigned short* __restrict__ hSelf, int ldSelf,
                                 unsigned short* __restrict__ hNext, int ldNext) {
  int idx = blockIdx.x * blockDim.x + threadIdx.x;
  if (idx >= B_ * H_) return;
  int b = idx >> 7, k = idx & 127;
  const float* gb = gates + (size_t)b * 512;
  float ig = sigm(gb[k]);
  float fg = sigm(gb[128 + k]);
  float gg = tanhf(gb[256 + k]);
  float og = sigm(gb[384 + k]);
  float c2 = fg * c[idx] + ig * gg;
  float h2 = og * tanhf(c2);
  c[idx] = c2;
  unsigned short hb = f2bf(h2);
  hSelf[(size_t)b * ldSelf + k] = hb;
  hNext[(size_t)b * ldNext + k] = hb;
}

// ------------------------------- attention ---------------------------------
// One block (64 threads) per batch row: q = ht@attn_w^T+b, windowed masked
// softmax over S=27, Gaussian reweight, ctx -> feat[:,0:131] (bf16),
// att -> d_out attention section.
__global__ void attn_kernel(const float* __restrict__ enc_out,
                            const unsigned char* __restrict__ sn_mask,
                            const int* __restrict__ sp_pos,
                            const float* __restrict__ attn_w,
                            const float* __restrict__ attn_b,
                            const unsigned short* __restrict__ feat_ht,  // feat base (ht at col 131)
                            unsigned short* __restrict__ feat_ctx,       // feat base (ctx at col 0)
                            float* __restrict__ att_out, int t) {
  __shared__ float ht[H_];
  __shared__ float q[131];
  __shared__ float sc[S_];
  __shared__ float red[2];
  int b = blockIdx.x;
  int tid = threadIdx.x;
  for (int k = tid; k < H_; k += 64)
    ht[k] = bf2f(feat_ht[(size_t)b * 288 + 131 + k]);
  __syncthreads();
  for (int j = tid; j < 131; j += 64) {
    float s = attn_b[j];
    const float* wr = attn_w + (size_t)j * H_;
    for (int k = 0; k < H_; ++k) s += ht[k] * wr[k];
    q[j] = s;
  }
  __syncthreads();
  int pos = sp_pos[(size_t)b * T_ + t];
  if (tid < S_) {
    const float* er = enc_out + ((size_t)b * S_ + tid) * 131;
    float s = 0.0f;
    for (int d = 0; d < 131; ++d) s += q[d] * er[d];
    int left  = pos - 1; if (left < 0) left = 0; if (left > S_ - 1) left = S_ - 1;
    int right = pos + 1; if (right > S_ - 1) right = S_ - 1; if (right < 0) right = 0;
    bool pad = (sn_mask[(size_t)b * S_ + tid] == 0);
    bool outside = (tid < left) || (tid > right);
    if (pad || outside) s += -1e9f;
    sc[tid] = s;
  }
  __syncthreads();
  if (tid == 0) {
    float m = sc[0];
    for (int s = 1; s < S_; ++s) m = fmaxf(m, sc[s]);
    red[0] = m;
  }
  __syncthreads();
  if (tid < S_) sc[tid] = __expf(sc[tid] - red[0]);
  __syncthreads();
  if (tid == 0) {
    float su = 0.0f;
    for (int s = 0; s < S_; ++s) su += sc[s];
    red[1] = su;
  }
  __syncthreads();
  if (tid < S_) {
    float d = (float)tid - (float)pos;
    float att = sc[tid] / red[1] * __expf(-2.0f * d * d);   // gauss_den = 0.5
    sc[tid] = att;
    att_out[((size_t)b * T_ + t) * S_ + tid] = att;
  }
  __syncthreads();
  for (int j = tid; j < 131; j += 64) {
    float cx = 0.0f;
    for (int s = 0; s < S_; ++s) cx += sc[s] * enc_out[((size_t)b * S_ + s) * 131 + j];
    feat_ctx[(size_t)b * 288 + j] = f2bf(cx);
  }
}

// ------------------------------- launcher ----------------------------------

extern "C" void kernel_launch(void* const* d_in, const int* in_sizes, int n_in,
                              void* d_out, int out_size, void* d_ws, size_t ws_size,
                              hipStream_t stream) {
  (void)in_sizes; (void)n_in; (void)out_size; (void)ws_size;
  const float*         dec_emb  = (const float*)d_in[0];
  const float*         enc_out  = (const float*)d_in[1];
  const unsigned char* sn_mask  = (const unsigned char*)d_in[2];
  const int*           sp_pos   = (const int*)d_in[3];
  const float*         w_ih1    = (const float*)d_in[4];
  const float*         w_ihR    = (const float*)d_in[5];
  const float*         w_hh     = (const float*)d_in[6];
  const float*         b_ih     = (const float*)d_in[7];
  const float*         b_hh     = (const float*)d_in[8];
  const float*         attn_w   = (const float*)d_in[9];
  const float*         attn_b   = (const float*)d_in[10];
  const float*         d1_w = (const float*)d_in[11]; const float* d1_b = (const float*)d_in[12];
  const float*         d2_w = (const float*)d_in[13]; const float* d2_b = (const float*)d_in[14];
  const float*         d3_w = (const float*)d_in[15]; const float* d3_b = (const float*)d_in[16];
  const float*         d4_w = (const float*)d_in[17]; const float* d4_b = (const float*)d_in[18];
  const float*         d5_w = (const float*)d_in[19]; const float* d5_b = (const float*)d_in[20];

  // workspace carve-out (256B aligned chunks)
  char* wsb = (char*)d_ws; size_t off = 0;
  auto alloc = [&](size_t bytes) -> void* {
    void* p = wsb + off; off = (off + bytes + 255) & ~(size_t)255; return p;
  };
  unsigned short* wcat0 = (unsigned short*)alloc((size_t)512 * 928 * 2);      // [w_ih1 | w_hh0] bf16
  unsigned short* wcatR = (unsigned short*)alloc((size_t)7 * 512 * 256 * 2);  // [w_ih | w_hh] l=1..7
  float*          biasc = (float*)alloc((size_t)8 * 512 * 4);                 // b_ih + b_hh
  unsigned short* wd1   = (unsigned short*)alloc((size_t)512 * 288 * 2);
  unsigned short* wd2   = (unsigned short*)alloc((size_t)256 * 512 * 2);
  unsigned short* wd3   = (unsigned short*)alloc((size_t)256 * 256 * 2);
  unsigned short* wd4   = (unsigned short*)alloc((size_t)256 * 256 * 2);
  unsigned short* wd5   = (unsigned short*)alloc((size_t)64 * 256 * 2);       // 51 rows + zero pad
  unsigned short* xin0  = (unsigned short*)alloc((size_t)B_ * 928 * 2);       // [x_t | h0_prev]
  unsigned short* xinR  = (unsigned short*)alloc((size_t)7 * B_ * 256 * 2);   // [h_{l-1,t} | h_{l,prev}]
  unsigned short* feat  = (unsigned short*)alloc((size_t)B_ * 288 * 2);       // [ctx(131) | ht(128) | pad]
  unsigned short* x1    = (unsigned short*)alloc((size_t)B_ * 512 * 2);
  unsigned short* x2    = (unsigned short*)alloc((size_t)B_ * 256 * 2);
  unsigned short* x3    = (unsigned short*)alloc((size_t)B_ * 256 * 2);
  unsigned short* x4    = (unsigned short*)alloc((size_t)B_ * 256 * 2);
  float*          gates = (float*)alloc((size_t)B_ * 512 * 4);
  float*          cst   = (float*)alloc((size_t)8 * B_ * H_ * 4);

  auto Z16 = [&](unsigned short* p, int n) {
    zero_u16_kernel<<<(n + 255) / 256, 256, 0, stream>>>(p, n);
  };
  auto Z32 = [&](float* p, int n) {
    zero_f32_kernel<<<(n + 255) / 256, 256, 0, stream>>>(p, n);
  };
  auto PK = [&](const float* s, unsigned short* d, int rows, int sK, int ld, int o2, int cK) {
    int tot = rows * cK;
    pack_sub_kernel<<<(tot + 255) / 256, 256, 0, stream>>>(s, d, rows, sK, ld, o2, cK);
  };

  // -------- prep (runs every call; deterministic; graph-capturable) --------
  Z16(wcat0, 512 * 928);
  Z16(wd1, 512 * 288);
  Z16(wd5, 64 * 256);
  Z16(xin0, B_ * 928);
  Z16(xinR, 7 * B_ * 256);
  Z16(feat, B_ * 288);
  Z32(cst, 8 * B_ * H_);
  PK(w_ih1, wcat0, 512, 770, 928, 0, 770);
  PK(w_hh, wcat0, 512, 128, 928, 770, 128);
  for (int l = 1; l < 8; ++l) {
    PK(w_ihR + (size_t)(l - 1) * 512 * 128, wcatR + (size_t)(l - 1) * 512 * 256, 512, 128, 256, 0, 128);
    PK(w_hh + (size_t)l * 512 * 128,        wcatR + (size_t)(l - 1) * 512 * 256, 512, 128, 256, 128, 128);
  }
  PK(d1_w, wd1, 512, 259, 288, 0, 259);
  PK(d2_w, wd2, 256, 512, 512, 0, 512);
  PK(d3_w, wd3, 256, 256, 256, 0, 256);
  PK(d4_w, wd4, 256, 256, 256, 0, 256);
  PK(d5_w, wd5, 51, 256, 256, 0, 256);
  bias_comb_kernel<<<(8 * 512 + 255) / 256, 256, 0, stream>>>(b_ih, b_hh, biasc, 8 * 512);

  float* loc_out = (float*)d_out;                            // [B, T, 51]
  float* att_out = (float*)d_out + (size_t)B_ * T_ * 51;     // [B, T, S]

  auto GEMM = [&](const unsigned short* X, int ldX, const unsigned short* W, int ldW,
                  const float* bias, float* Yf, int ldYf, unsigned short* Yb, int ldYb,
                  int Mt2, int Nt2, int Nv, int Kp, int relu) {
    wmma_gemm_kernel<<<(Mt2 * Nt2) / 8, 256, 0, stream>>>(X, ldX, W, ldW, bias, Yf, ldYf,
                                                          Yb, ldYb, Mt2, Nt2, Nv, Kp, relu);
  };

  // -------------------------- time recurrence ------------------------------
  for (int t = 0; t < T_; ++t) {
    convert_x_kernel<<<(B_ * 770 + 255) / 256, 256, 0, stream>>>(dec_emb, xin0, t);
    for (int l = 0; l < 8; ++l) {
      const unsigned short* X = (l == 0) ? xin0 : xinR + (size_t)(l - 1) * B_ * 256;
      const unsigned short* W = (l == 0) ? wcat0 : wcatR + (size_t)(l - 1) * 512 * 256;
      int Kp = (l == 0) ? 928 : 256;
      GEMM(X, Kp, W, Kp, biasc + (size_t)l * 512, gates, 512, nullptr, 0, 32, 16, 512, Kp, 0);
      unsigned short* hSelf = (l == 0) ? xin0 + 770 : xinR + (size_t)(l - 1) * B_ * 256 + 128;
      int ldSelf = (l == 0) ? 928 : 256;
      unsigned short* hNext; int ldNext;
      if (l < 7) { hNext = xinR + (size_t)l * B_ * 256; ldNext = 256; }
      else       { hNext = feat + 131;                  ldNext = 288; }
      lstm_gate_kernel<<<(B_ * H_ + 255) / 256, 256, 0, stream>>>(
          gates, cst + (size_t)l * B_ * H_, hSelf, ldSelf, hNext, ldNext);
    }
    attn_kernel<<<B_, 64, 0, stream>>>(enc_out, sn_mask, sp_pos, attn_w, attn_b,
                                       feat, feat, att_out, t);
    GEMM(feat, 288, wd1, 288, d1_b, nullptr, 0, x1, 512, 32, 16, 512, 288, 1);
    GEMM(x1, 512, wd2, 512, d2_b, nullptr, 0, x2, 256, 32, 8, 256, 512, 1);
    GEMM(x2, 256, wd3, 256, d3_b, nullptr, 0, x3, 256, 32, 8, 256, 256, 1);
    GEMM(x3, 256, wd4, 256, d4_b, nullptr, 0, x4, 256, 32, 8, 256, 256, 1);
    GEMM(x4, 256, wd5, 256, d5_b, loc_out + (size_t)t * 51, T_ * 51, nullptr, 0, 32, 2, 51, 256, 0);
  }
}
